// PTv3KD_27006754357920
// MI455X (gfx1250) — matrix operand order repaired and checked by reading, not compiled, not run on hardware
//
#include <hip/hip_runtime.h>
#include <hip/hip_bf16.h>
#include <cstdint>
#include <cstddef>

typedef _Float16 half_t;
typedef __attribute__((ext_vector_type(16))) _Float16 v16h;
typedef __attribute__((ext_vector_type(8)))  _Float16 v8h;
typedef __attribute__((ext_vector_type(8)))  float    v8f;

#if defined(__AMDGCN__) && __has_builtin(__builtin_amdgcn_global_load_async_to_lds_b128) && \
    __has_builtin(__builtin_amdgcn_s_wait_asynccnt)
#define HAVE_ASYNC_LDS 1
#else
#define HAVE_ASYNC_LDS 0
#endif

#if HAVE_ASYNC_LDS
typedef int i128t __attribute__((__vector_size__(16)));
#define ASYNC_GPTR(x) ((__attribute__((address_space(1))) i128t*)(x))
#define ASYNC_LPTR(x) ((__attribute__((address_space(3))) i128t*)(x))
#endif

static inline int pad32(int x) { return (x + 31) & ~31; }

// ---------------------------------------------------------------------------
// Device helpers
// ---------------------------------------------------------------------------
__device__ __forceinline__ float geluf(float x) {
  float x3 = x * x * x;
  return 0.5f * x * (1.f + tanhf(0.7978845608028654f * (x + 0.044715f * x3)));
}

// ---------------------------------------------------------------------------
// Weight / bias preparation (per-launch; tiny).  Weights stored TRANSPOSED:
// WT[m * Kp + k], zero-padded, so B-fragments are contiguous 16B loads.
// ---------------------------------------------------------------------------
__global__ void k_prep_w(const float* __restrict__ w, half_t* __restrict__ wt,
                         int K, int M, int Kp, int Mp) {
  int i = blockIdx.x * blockDim.x + threadIdx.x;
  if (i < Kp * Mp) {
    int m = i / Kp, k = i - m * Kp;
    wt[i] = (k < K && m < M) ? (half_t)w[(size_t)k * M + m] : (half_t)0.0f;
  }
}
__global__ void k_prep_b(const float* __restrict__ b, float* __restrict__ bp,
                         int M, int Mp) {
  int i = blockIdx.x * blockDim.x + threadIdx.x;
  if (i < Mp) bp[i] = (i < M) ? b[i] : 0.f;
}
// qkv weight: remap output cols to (s, h, j<32) padded-per-head layout
__global__ void k_prep_w_qkv(const float* __restrict__ w, half_t* __restrict__ wt,
                             int K, int dim, int heads, int dh, int Kp) {
  int H32 = heads * 32, Mp = 3 * H32;
  int i = blockIdx.x * blockDim.x + threadIdx.x;
  if (i < Kp * Mp) {
    int m = i / Kp, k = i - m * Kp;
    int s = m / H32, r = m - s * H32, h = r >> 5, j = r & 31;
    wt[i] = (k < K && j < dh)
                ? (half_t)w[(size_t)k * (3 * dim) + s * dim + h * dh + j]
                : (half_t)0.0f;
  }
}
__global__ void k_prep_b_qkv(const float* __restrict__ b, float* __restrict__ bp,
                             int dim, int heads, int dh) {
  int H32 = heads * 32, Mp = 3 * H32;
  int i = blockIdx.x * blockDim.x + threadIdx.x;
  if (i < Mp) {
    int s = i / H32, r = i - s * H32, h = r >> 5, j = r & 31;
    bp[i] = (j < dh) ? b[s * dim + h * dh + j] : 0.f;
  }
}
// proj weight: remap input rows from packed (h*dh+j) to padded (h*32+j)
__global__ void k_prep_w_proj(const float* __restrict__ w, half_t* __restrict__ wt,
                              int dim, int heads, int dh, int Mp) {
  int Kp = heads * 32;
  int i = blockIdx.x * blockDim.x + threadIdx.x;
  if (i < Kp * Mp) {
    int m = i / Kp, k = i - m * Kp;
    int h = k >> 5, j = k & 31;
    wt[i] = (j < dh && m < dim) ? (half_t)w[(size_t)(h * dh + j) * dim + m]
                                : (half_t)0.0f;
  }
}

// ---------------------------------------------------------------------------
// Elementwise utility kernels (all write full padded rows; pads = 0)
// ---------------------------------------------------------------------------
__global__ void k_coordf(const int* __restrict__ c, half_t* __restrict__ d,
                         int step, int nOut) {
  int i = blockIdx.x * blockDim.x + threadIdx.x;
  if (i < nOut * 32) {
    int r = i >> 5, j = i & 31;
    d[i] = (j < 3) ? (half_t)(float)c[(size_t)r * step * 3 + j] : (half_t)0.0f;
  }
}
__global__ void k_padf(const float* __restrict__ s, half_t* __restrict__ d,
                       int n, int dreal, int dp) {
  int i = blockIdx.x * blockDim.x + threadIdx.x;
  if (i < n * dp) {
    int r = i / dp, c = i - r * dp;
    d[i] = (c < dreal) ? (half_t)s[(size_t)r * dreal + c] : (half_t)0.0f;
  }
}
// pair-mean pooling over full padded width (pads are zero and stay zero)
__global__ void k_pool(const float* __restrict__ x, float* __restrict__ yf,
                       half_t* __restrict__ yh, int nOut, int dp) {
  int i = blockIdx.x * blockDim.x + threadIdx.x;
  if (i < nOut * dp) {
    int r = i / dp, c = i - r * dp;
    float v = 0.5f * (x[(size_t)(2 * r) * dp + c] + x[(size_t)(2 * r + 1) * dp + c]);
    if (yf) yf[i] = v;
    yh[i] = (half_t)v;
  }
}
// out[n] = [a[n] (da of sa), b[n>>1] (db of sb), zeros] with stride so
__global__ void k_concat(const half_t* __restrict__ a, int da, int sa,
                         const half_t* __restrict__ b, int db, int sb,
                         half_t* __restrict__ o, int so, int n) {
  int i = blockIdx.x * blockDim.x + threadIdx.x;
  if (i < n * so) {
    int r = i / so, c = i - r * so;
    half_t v = (half_t)0.0f;
    if (c < da) v = a[(size_t)r * sa + c];
    else if (c < da + db) v = b[(size_t)(r >> 1) * sb + (c - da)];
    o[i] = v;
  }
}

// one wave per row; pads written as zero
__global__ __launch_bounds__(256) void k_layernorm(
    const float* __restrict__ x, const float* __restrict__ g, const float* __restrict__ b,
    float* __restrict__ yf, half_t* __restrict__ yh, int N, int d, int dp) {
  int row = blockIdx.x * (blockDim.x >> 5) + (threadIdx.x >> 5);
  if (row >= N) return;
  int lane = threadIdx.x & 31;
  const float* xr = x + (size_t)row * dp;
  float s = 0.f, s2 = 0.f;
  for (int i = lane; i < d; i += 32) { float v = xr[i]; s += v; s2 += v * v; }
  for (int m = 16; m > 0; m >>= 1) { s += __shfl_xor(s, m, 32); s2 += __shfl_xor(s2, m, 32); }
  float mean = s / (float)d;
  float inv = rsqrtf(s2 / (float)d - mean * mean + 1e-5f);
  for (int i = lane; i < dp; i += 32) {
    float v = (i < d) ? (xr[i] - mean) * inv * g[i] + b[i] : 0.f;
    size_t off = (size_t)row * dp + i;
    if (yf) yf[off] = v;
    if (yh) yh[off] = (half_t)v;
  }
}

// ---------------------------------------------------------------------------
// Fused linear on fully-padded operands:  Y = [act](X @ W + b) [+ res].
// X: N x Kp (f16, Kp % 32 == 0, pads zero).  WT: Mp x Kp transposed f16.
// One wave = one 16x16 tile; all fragment loads are unconditional b128.
// ---------------------------------------------------------------------------
__global__ __launch_bounds__(256) void k_linear(
    const half_t* __restrict__ X, const half_t* __restrict__ WT,
    const float* __restrict__ biasP, const float* __restrict__ res,
    float* __restrict__ Yf, half_t* __restrict__ Yh,
    int N, int Kp, int Mp, int Mreal, int strideO, int act) {
  const int lane = threadIdx.x & 31;
  const int wave = threadIdx.x >> 5;
  const int mt = Mp >> 4;
  const long long tile = (long long)blockIdx.x * 8 + wave;
  const long long nt = (long long)(N >> 4) * mt;
  if (tile >= nt) return;                 // wave-uniform
  const int rowTile = (int)(tile / mt);
  const int colTile = (int)(tile % mt);
  const int half_id = lane >> 4;
  const int lr = lane & 15;
  const half_t* __restrict__ xrow = X + (size_t)(rowTile * 16 + lr) * Kp;
  const half_t* __restrict__ wcol = WT + (size_t)(colTile * 16 + lr) * Kp;
  v8f acc = {0.f, 0.f, 0.f, 0.f, 0.f, 0.f, 0.f, 0.f};
  for (int k0 = 0; k0 < Kp; k0 += 32) {
    v8h alo = *(const v8h*)(xrow + k0 + half_id * 8);
    v8h ahi = *(const v8h*)(xrow + k0 + 16 + half_id * 8);
    v8h blo = *(const v8h*)(wcol + k0 + half_id * 8);
    v8h bhi = *(const v8h*)(wcol + k0 + 16 + half_id * 8);
    v16h a, b;
#pragma unroll
    for (int i = 0; i < 8; ++i) {
      a[i] = alo[i]; a[i + 8] = ahi[i];
      b[i] = blo[i]; b[i + 8] = bhi[i];
    }
    __builtin_prefetch(xrow + k0 + 64, 0, 1);
    acc = __builtin_amdgcn_wmma_f32_16x16x32_f16(false, a, false, b, (short)0, acc,
                                                 false, false);
  }
  const int bcol = colTile * 16 + lr;
  if (bcol < strideO) {
    const float bv = biasP[bcol];
#pragma unroll
    for (int i = 0; i < 8; ++i) {
      int r = rowTile * 16 + half_id * 8 + i;
      size_t off = (size_t)r * strideO + bcol;
      float v;
      if (bcol < Mreal) {
        v = acc[i] + bv;
        if (act) v = geluf(v);
        if (res) v += res[off];
      } else {
        v = 0.f;                          // keep pad columns exactly zero
      }
      if (Yf) Yf[off] = v;
      if (Yh) Yh[off] = (half_t)v;
    }
  }
}

// ---------------------------------------------------------------------------
// Fused patch attention on per-head-padded qkv: layout (n, 3, heads, 32),
// stride S = 3*heads*32, pads zero.  One wave = (patch, head, 16-row block);
// the 4 waves of a block share the same (patch, head), so K is staged to LDS
// once per block with GLOBAL_LOAD_ASYNC_TO_LDS (ASYNCcnt), and V is staged
// TRANSPOSED so both K- and V-fragments are ds_load_b128 pairs.
// ---------------------------------------------------------------------------
__global__ __launch_bounds__(128) void k_attn(
    const half_t* __restrict__ qkv, half_t* __restrict__ o, int heads, int dh) {
  __shared__ __align__(16) half_t Ksh[256 * 32];       // 16 KB  K[t][d]
  __shared__ __align__(16) half_t VTsh[32 * 256];      // 16 KB  V^T[d][t]
  __shared__ __align__(16) half_t Psh[4 * 16 * 256];   // 32 KB  per-wave P
  const int H32 = heads * 32, S = 3 * H32;
  const int lane = threadIdx.x & 31;
  const int wave = threadIdx.x >> 5;
  const int gw = blockIdx.x * 4 + wave;
  const int rb = gw & 15;
  const int hp = gw >> 4;                // identical for all 4 waves of a block
  const int h = hp % heads;
  const int p = hp / heads;
  const int half_id = lane >> 4;
  const int lr = lane & 15;
  const size_t base = (size_t)p * 256;
  const float scale = rsqrtf((float)dh);

  // ---- cooperative staging: K (async linear copy) + V (manual transpose) ----
  {
    const half_t* kbase = qkv + base * S + H32 + h * 32;
    const half_t* vbase = qkv + base * S + 2 * H32 + h * 32;
    for (int c = threadIdx.x; c < 1024; c += 128) {    // 1024 x 16B chunks each
      int t = c >> 2, q = c & 3;
#if HAVE_ASYNC_LDS
      __builtin_amdgcn_global_load_async_to_lds_b128(
          ASYNC_GPTR(kbase + (size_t)t * S + q * 8),
          ASYNC_LPTR(&Ksh[t * 32 + q * 8]), 0, 0);
#else
      *(v8h*)&Ksh[t * 32 + q * 8] = *(const v8h*)(kbase + (size_t)t * S + q * 8);
#endif
      v8h vv = *(const v8h*)(vbase + (size_t)t * S + q * 8);
#pragma unroll
      for (int j = 0; j < 8; ++j) VTsh[(q * 8 + j) * 256 + t] = vv[j];
    }
#if HAVE_ASYNC_LDS
    __builtin_amdgcn_s_wait_asynccnt(0);
#endif
    __syncthreads();
  }

  // ---- Q A-fragment: two unconditional 16B global loads ----
  v16h qa;
  {
    const half_t* qrow = qkv + (base + rb * 16 + lr) * S + h * 32;
    v8h lo = *(const v8h*)(qrow + half_id * 8);
    v8h hi = *(const v8h*)(qrow + 16 + half_id * 8);
#pragma unroll
    for (int i = 0; i < 8; ++i) { qa[i] = lo[i]; qa[i + 8] = hi[i]; }
  }

  // ---- S = Q K^T : 16 tiles kept in accumulators; K from LDS ----
  v8f acc[16];
#pragma unroll
  for (int ct = 0; ct < 16; ++ct) {
    const half_t* krow = &Ksh[(ct * 16 + lr) * 32];
    v8h lo = *(const v8h*)(krow + half_id * 8);
    v8h hi = *(const v8h*)(krow + 16 + half_id * 8);
    v16h kb;
#pragma unroll
    for (int i = 0; i < 8; ++i) { kb[i] = lo[i]; kb[i + 8] = hi[i]; }
    v8f z = {0.f, 0.f, 0.f, 0.f, 0.f, 0.f, 0.f, 0.f};
    acc[ct] = __builtin_amdgcn_wmma_f32_16x16x32_f16(false, qa, false, kb, (short)0, z,
                                                     false, false);
  }

  // ---- scale + row softmax (rows r = half_id*8 + i, cols across 16 lanes) ----
  float inv[8];
#pragma unroll
  for (int i = 0; i < 8; ++i) {
    float m = -3.0e38f;
#pragma unroll
    for (int ct = 0; ct < 16; ++ct) { acc[ct][i] *= scale; m = fmaxf(m, acc[ct][i]); }
    for (int s = 1; s < 16; s <<= 1) m = fmaxf(m, __shfl_xor(m, s, 32));
    float sum = 0.f;
#pragma unroll
    for (int ct = 0; ct < 16; ++ct) { float e = __expf(acc[ct][i] - m); acc[ct][i] = e; sum += e; }
    for (int s = 1; s < 16; s <<= 1) sum += __shfl_xor(sum, s, 32);
    inv[i] = 1.f / sum;
  }

  // ---- stage normalized P (16x256 f16) into this wave's LDS slice ----
  const int wbase = wave * 16 * 256;
#pragma unroll
  for (int ct = 0; ct < 16; ++ct)
#pragma unroll
    for (int i = 0; i < 8; ++i)
      Psh[wbase + (half_id * 8 + i) * 256 + ct * 16 + lr] = (half_t)(acc[ct][i] * inv[i]);

  // ---- O = P @ V : both fragments from LDS (ds_load_b128 pairs) ----
#pragma unroll
  for (int oc = 0; oc < 2; ++oc) {
    v8f ao = {0.f, 0.f, 0.f, 0.f, 0.f, 0.f, 0.f, 0.f};
    const int d = oc * 16 + lr;
#pragma unroll
    for (int kt = 0; kt < 8; ++kt) {
      v8h plo = *(const v8h*)&Psh[wbase + lr * 256 + kt * 32 + half_id * 8];
      v8h phi = *(const v8h*)&Psh[wbase + lr * 256 + kt * 32 + 16 + half_id * 8];
      v16h pa;
#pragma unroll
      for (int i = 0; i < 8; ++i) { pa[i] = plo[i]; pa[i + 8] = phi[i]; }
      v8h vlo = *(const v8h*)&VTsh[d * 256 + kt * 32 + half_id * 8];
      v8h vhi = *(const v8h*)&VTsh[d * 256 + kt * 32 + 16 + half_id * 8];
      v16h vb;
#pragma unroll
      for (int i = 0; i < 8; ++i) { vb[i] = vlo[i]; vb[i + 8] = vhi[i]; }
      ao = __builtin_amdgcn_wmma_f32_16x16x32_f16(false, pa, false, vb, (short)0, ao,
                                                  false, false);
    }
#pragma unroll
    for (int i = 0; i < 8; ++i) {
      size_t row = base + rb * 16 + half_id * 8 + i;
      o[row * H32 + h * 32 + d] = (half_t)ao[i];
    }
  }
}

// ---------------------------------------------------------------------------
// Host-side orchestration
// ---------------------------------------------------------------------------
namespace {

struct Bump {
  char* base; size_t off, cap;
  void* take(size_t bytes) {
    size_t a = (off + 255) & ~(size_t)255;
    off = a + bytes;
    return base + a;
  }
  float*  f(size_t n) { return (float*)take(n * 4); }
  half_t* h(size_t n) { return (half_t*)take(n * 2); }
};

struct PWalk { void* const* in; int idx; };
inline const float* nxt(PWalk& p) { return (const float*)p.in[p.idx++]; }

struct Lin { const float* b; const float* w; int din, dout; };
struct LNp { const float* b; const float* g; };
struct Blk { Lin cpe1, cpe2, fc1, fc2; LNp ln1, ln2; Lin proj, qkv; };

// jax flattens dicts sorted by key: _lin_p {'b','w'}, _ln_p {'b','g'}
inline Lin getLin(PWalk& p, int din, int dout) {
  Lin l; l.b = nxt(p); l.w = nxt(p); l.din = din; l.dout = dout; return l;
}
inline LNp getLN(PWalk& p) { LNp l; l.b = nxt(p); l.g = nxt(p); return l; }
// block keys sorted: cpe1, cpe2, fc1, fc2, ln1, ln2, proj, qkv
inline Blk getBlk(PWalk& p, int dim) {
  Blk b;
  b.cpe1 = getLin(p, 3, dim);
  b.cpe2 = getLin(p, dim, dim);
  b.fc1  = getLin(p, dim, 3 * dim);
  b.fc2  = getLin(p, 3 * dim, dim);
  b.ln1  = getLN(p);
  b.ln2  = getLN(p);
  b.proj = getLin(p, dim, dim);
  b.qkv  = getLin(p, dim, 3 * dim);
  return b;
}

inline int ceilDiv(long long a, long long b) { return (int)((a + b - 1) / b); }

inline void launchLin(hipStream_t s, const half_t* X, const half_t* WT, const float* bp,
                      const float* res, float* Yf, half_t* Yh,
                      int N, int Kp, int Mp, int Mreal, int strideO, int act) {
  long long tiles = (long long)(N >> 4) * (Mp >> 4);
  k_linear<<<ceilDiv(tiles, 8), 256, 0, s>>>(X, WT, bp, res, Yf, Yh,
                                             N, Kp, Mp, Mreal, strideO, act);
}

// generic linear: Kp = pad32(din) matches producer strides by construction
inline void linG(hipStream_t s, Bump& B, const half_t* X, const Lin& L,
                 const float* res, float* Yf, half_t* Yh, int N, int act,
                 int strideO_override = -1) {
  int Kp = pad32(L.din), Mp = pad32(L.dout);
  int strideO = strideO_override > 0 ? strideO_override : Mp;
  half_t* WT = B.h((size_t)Kp * Mp);
  float* bp = B.f(Mp);
  k_prep_w<<<ceilDiv((long long)Kp * Mp, 256), 256, 0, s>>>(L.w, WT, L.din, L.dout, Kp, Mp);
  k_prep_b<<<ceilDiv(Mp, 256), 256, 0, s>>>(L.b, bp, L.dout, Mp);
  launchLin(s, X, WT, bp, res, Yf, Yh, N, Kp, Mp, L.dout, strideO, act);
}

inline void linQKV(hipStream_t s, Bump& B, const half_t* X, const Lin& L,
                   half_t* Yh, int N, int dim, int heads) {
  int dh = dim / heads, H32 = heads * 32;
  int Kp = pad32(dim), Mp = 3 * H32;
  half_t* WT = B.h((size_t)Kp * Mp);
  float* bp = B.f(Mp);
  k_prep_w_qkv<<<ceilDiv((long long)Kp * Mp, 256), 256, 0, s>>>(L.w, WT, dim, dim, heads, dh, Kp);
  k_prep_b_qkv<<<ceilDiv(Mp, 256), 256, 0, s>>>(L.b, bp, dim, heads, dh);
  launchLin(s, X, WT, bp, nullptr, nullptr, Yh, N, Kp, Mp, Mp, Mp, 0);
}

inline void linProj(hipStream_t s, Bump& B, const half_t* X, const Lin& L,
                    float* xf, half_t* xh, int N, int dim, int heads) {
  int dh = dim / heads, H32 = heads * 32;
  int Kp = H32, Mp = pad32(dim);
  half_t* WT = B.h((size_t)Kp * Mp);
  float* bp = B.f(Mp);
  k_prep_w_proj<<<ceilDiv((long long)Kp * Mp, 256), 256, 0, s>>>(L.w, WT, dim, heads, dh, Mp);
  k_prep_b<<<ceilDiv(Mp, 256), 256, 0, s>>>(L.b, bp, dim, Mp);
  launchLin(s, X, WT, bp, xf, xf, xh, N, Kp, Mp, dim, Mp, 0);
}

inline void lnL(hipStream_t s, const float* x, const LNp& L, float* yf, half_t* yh,
                int N, int d, int dp) {
  k_layernorm<<<ceilDiv(N, 8), 256, 0, s>>>(x, L.g, L.b, yf, yh, N, d, dp);
}

inline void attnL(hipStream_t s, const half_t* qkv, half_t* o, int Np, int dim, int heads) {
  int waves = (Np / 256) * heads * 16;          // always divisible by 4
  k_attn<<<waves / 4, 128, 0, s>>>(qkv, o, heads, dim / heads);
}

struct Tmp {
  half_t *tq, *tm, *th, *to, *tc, *tcat, *tph;
  float* txf;
};

inline void runBlock(hipStream_t s, Bump& B, const Blk& bk, float* xf, half_t* xh,
                     const half_t* cf, int Np, int dim, int heads, Tmp& T) {
  const int dp = pad32(dim);
  // x += cpe2(gelu(cpe1(coordf)))
  linG(s, B, cf, bk.cpe1, nullptr, nullptr, T.tc, Np, 1);
  linG(s, B, T.tc, bk.cpe2, xf, xf, xh, Np, 0);
  // attention branch
  lnL(s, xf, bk.ln1, nullptr, T.th, Np, dim, dp);
  linQKV(s, B, T.th, bk.qkv, T.tq, Np, dim, heads);
  attnL(s, T.tq, T.to, Np, dim, heads);
  linProj(s, B, T.to, bk.proj, xf, xh, Np, dim, heads);
  // MLP branch
  lnL(s, xf, bk.ln2, nullptr, T.th, Np, dim, dp);
  linG(s, B, T.th, bk.fc1, nullptr, nullptr, T.tm, Np, 1);
  linG(s, B, T.tm, bk.fc2, xf, xf, xh, Np, 0);
}

} // namespace

extern "C" void kernel_launch(void* const* d_in, const int* in_sizes, int n_in,
                              void* d_out, int out_size, void* d_ws, size_t ws_size,
                              hipStream_t stream) {
  (void)in_sizes; (void)n_in; (void)out_size;
  const int* coord = (const int*)d_in[0];
  const float* feat = (const float*)d_in[1];

  // ---- params, flattened in jax sorted-key order ----
  PWalk p{d_in, 2};
  Blk bott  = getBlk(p, 192);
  Lin cls   = getLin(p, 24, 13);
  Blk dec0  = getBlk(p, 192);
  Blk dec1  = getBlk(p, 96);
  Blk dec2  = getBlk(p, 48);
  Blk dec3  = getBlk(p, 24);
  Lin dp0   = getLin(p, 384, 192);
  Lin dp1   = getLin(p, 288, 96);
  Lin dp2   = getLin(p, 144, 48);
  Lin dp3   = getLin(p, 72, 24);
  Lin down1 = getLin(p, 24, 48);  LNp down1ln = getLN(p);
  Lin down2 = getLin(p, 48, 96);  LNp down2ln = getLN(p);
  Lin down3 = getLin(p, 96, 192); LNp down3ln = getLN(p);
  Blk enc0  = getBlk(p, 24);
  Blk enc1  = getBlk(p, 48);
  Blk enc2a = getBlk(p, 96);
  Blk enc2b = getBlk(p, 96);
  Blk enc2c = getBlk(p, 96);
  Blk enc3  = getBlk(p, 192);
  LNp headln = getLN(p);
  Lin stem   = getLin(p, 6, 24);  LNp stemln = getLN(p);

  Bump B{(char*)d_ws, 0, ws_size};
  const int N0 = 131072, N1 = 65536, N2 = 32768, N3 = 16384, N4 = 8192;
  // padded dims: 24->32, 48->64, 96->96, 192->192

  // ---- coords (f16, stride 32, pads zero) per level ----
  half_t* cf[5];
  const int Ns[5] = {N0, N1, N2, N3, N4};
  for (int l = 0; l < 5; ++l) {
    cf[l] = B.h((size_t)Ns[l] * 32);
    k_coordf<<<ceilDiv((long long)Ns[l] * 32, 256), 256, 0, stream>>>(coord, cf[l], 1 << l, Ns[l]);
  }

  // ---- shared temps (max over levels) ----
  Tmp T;
  T.tq   = B.h(25165824);  // max N * 3*heads*32   (131072*192)
  T.tm   = B.h(12582912);  // max N * pad32(3d)    (131072*96)
  T.th   = B.h(4194304);   // max N * dp           (131072*32)
  T.to   = B.h(8388608);   // max N * heads*32     (131072*64)
  T.tc   = B.h(4194304);
  T.tph  = B.h(2097152);   // max pooled N * dp_prev
  T.txf  = B.f(4194304);
  T.tcat = T.tq;           // alias: concat consumed before tq is written
  half_t* tfeat = T.tm;    // alias: consumed by stem before tm is written

  // ---- per-level persistent activations (f32 master + f16 shadow) ----
  float*  x0 = B.f((size_t)N0 * 32);  half_t* x0h = B.h((size_t)N0 * 32);
  float*  x1 = B.f((size_t)N1 * 64);  half_t* x1h = B.h((size_t)N1 * 64);
  float*  x2 = B.f((size_t)N2 * 96);  half_t* x2h = B.h((size_t)N2 * 96);
  float*  x3 = B.f((size_t)N3 * 192); half_t* x3h = B.h((size_t)N3 * 192);
  float*  x4 = B.f((size_t)N4 * 192); half_t* x4h = B.h((size_t)N4 * 192);
  float*  y3 = B.f((size_t)N3 * 192); half_t* y3h = B.h((size_t)N3 * 192);
  float*  y2 = B.f((size_t)N2 * 96);  half_t* y2h = B.h((size_t)N2 * 96);
  float*  y1 = B.f((size_t)N1 * 64);  half_t* y1h = B.h((size_t)N1 * 64);
  float*  y0 = B.f((size_t)N0 * 32);  half_t* y0h = B.h((size_t)N0 * 32);

  // ================= encoder =================
  k_padf<<<ceilDiv((long long)N0 * 32, 256), 256, 0, stream>>>(feat, tfeat, N0, 6, 32);
  linG(stream, B, tfeat, stem, nullptr, T.txf, nullptr, N0, 0);
  lnL(stream, T.txf, stemln, x0, x0h, N0, 24, 32);
  runBlock(stream, B, enc0, x0, x0h, cf[0], N0, 24, 2, T);

  k_pool<<<ceilDiv((long long)N1 * 32, 256), 256, 0, stream>>>(x0, nullptr, T.tph, N1, 32);
  linG(stream, B, T.tph, down1, nullptr, T.txf, nullptr, N1, 0);
  lnL(stream, T.txf, down1ln, x1, x1h, N1, 48, 64);
  runBlock(stream, B, enc1, x1, x1h, cf[1], N1, 48, 4, T);

  k_pool<<<ceilDiv((long long)N2 * 64, 256), 256, 0, stream>>>(x1, nullptr, T.tph, N2, 64);
  linG(stream, B, T.tph, down2, nullptr, T.txf, nullptr, N2, 0);
  lnL(stream, T.txf, down2ln, x2, x2h, N2, 96, 96);
  runBlock(stream, B, enc2a, x2, x2h, cf[2], N2, 96, 4, T);
  runBlock(stream, B, enc2b, x2, x2h, cf[2], N2, 96, 4, T);
  runBlock(stream, B, enc2c, x2, x2h, cf[2], N2, 96, 4, T);

  k_pool<<<ceilDiv((long long)N3 * 96, 256), 256, 0, stream>>>(x2, nullptr, T.tph, N3, 96);
  linG(stream, B, T.tph, down3, nullptr, T.txf, nullptr, N3, 0);
  lnL(stream, T.txf, down3ln, x3, x3h, N3, 192, 192);
  runBlock(stream, B, enc3, x3, x3h, cf[3], N3, 192, 8, T);

  // ================= bottleneck =================
  k_pool<<<ceilDiv((long long)N4 * 192, 256), 256, 0, stream>>>(x3, x4, x4h, N4, 192);
  runBlock(stream, B, bott, x4, x4h, cf[4], N4, 192, 8, T);

  // ================= decoder =================
  k_concat<<<ceilDiv((long long)N3 * 384, 256), 256, 0, stream>>>(x3h, 192, 192, x4h, 192, 192,
                                                                  T.tcat, 384, N3);
  linG(stream, B, T.tcat, dp0, nullptr, y3, y3h, N3, 0);
  runBlock(stream, B, dec0, y3, y3h, cf[3], N3, 192, 8, T);

  k_concat<<<ceilDiv((long long)N2 * 288, 256), 256, 0, stream>>>(x2h, 96, 96, y3h, 192, 192,
                                                                  T.tcat, 288, N2);
  linG(stream, B, T.tcat, dp1, nullptr, y2, y2h, N2, 0);
  runBlock(stream, B, dec1, y2, y2h, cf[2], N2, 96, 4, T);

  k_concat<<<ceilDiv((long long)N1 * 160, 256), 256, 0, stream>>>(x1h, 48, 64, y2h, 96, 96,
                                                                  T.tcat, 160, N1);
  linG(stream, B, T.tcat, dp2, nullptr, y1, y1h, N1, 0);
  runBlock(stream, B, dec2, y1, y1h, cf[1], N1, 48, 4, T);

  k_concat<<<ceilDiv((long long)N0 * 96, 256), 256, 0, stream>>>(x0h, 24, 32, y1h, 48, 64,
                                                                 T.tcat, 96, N0);
  linG(stream, B, T.tcat, dp3, nullptr, y0, y0h, N0, 0);
  runBlock(stream, B, dec3, y0, y0h, cf[0], N0, 24, 2, T);

  // ================= head =================
  lnL(stream, y0, headln, nullptr, T.th, N0, 24, 32);
  linG(stream, B, T.th, cls, nullptr, (float*)d_out, nullptr, N0, 0, /*strideO=*/13);
}
